// VectorQuantizer_1443109012075
// MI455X (gfx1250) — matrix operand order repaired and checked by reading, not compiled, not run on hardware
//
#include <hip/hip_runtime.h>
#include <hip/hip_bf16.h>

// Problem constants (from reference): B=16, D=256, H=64, W=64, K=1024
#define KCODES   1024
#define DDIM     256
#define OUT_ELEMS 16777216   // 16*256*64*64
#define NPIX     65536       // 16*64*64
#define NWG      512         // each WG handles 128 pixels (2 h-rows x 64 w)

// Padded row length (bf16 elems): 256 + 8 -> row stride 528 B (4-bank shift/row)
#define DPAD 264
#define ROWB 528            // DPAD * 2 bytes

// LDS layout (bytes) -- double-buffered E stage
#define XHI_OFF   0
#define XLO_OFF   67584     // 128*528
#define EB_OFF    135168    // E buffers: 2 x (hi 33792 + lo 33792)
#define EB_STRIDE 67584     // per buffer (hi+lo)
#define EHL_REL   33792     // lo region relative to hi within a buffer
#define XN_OFF    270336    // 128*4 = 512 B
#define MK_OFF    270848    // 128*8 = 1024 B
#define SMEM_BYTES 271872   // < 320 KB WGP LDS

typedef __attribute__((ext_vector_type(16))) __bf16 v16bf;
typedef __attribute__((ext_vector_type(8)))  __bf16 v8bf;
typedef __attribute__((ext_vector_type(8)))  float  v8f;

__device__ __forceinline__ unsigned short f2bf_bits(float f) {
  __bf16 b = (__bf16)f;
  return __builtin_bit_cast(unsigned short, b);
}
__device__ __forceinline__ float bf_bits2f(unsigned short u) {
  __bf16 b = __builtin_bit_cast(__bf16, u);
  return (float)b;
}
// Assemble a 16-element bf16 WMMA operand from two 16-byte LDS chunks
__device__ __forceinline__ v16bf ld_pair(const char* p0, const char* p1) {
  union { v16bf v; v8bf h[2]; } u;
  u.h[0] = *(const v8bf*)p0;
  u.h[1] = *(const v8bf*)p1;
  return u.v;
}

// ---------------- Kernel A: codebook prep (bf16 hi/lo split + ||e||^2) -----
__global__ __launch_bounds__(256) void vq_prep(const float* __restrict__ emb,
                                               unsigned short* __restrict__ ehi,
                                               unsigned short* __restrict__ elo,
                                               float* __restrict__ enorm) {
  __shared__ float red[256];
  const int k = blockIdx.x;     // 0..1023
  const int d = threadIdx.x;    // 0..255
  const float v  = emb[k * DDIM + d];
  const unsigned short hb = f2bf_bits(v);
  const float fh = bf_bits2f(hb);
  const unsigned short lb = f2bf_bits(v - fh);
  ehi[k * DDIM + d] = hb;
  elo[k * DDIM + d] = lb;
  red[d] = v * v;
  __syncthreads();
  #pragma unroll
  for (int s = 128; s > 0; s >>= 1) {
    if (d < s) red[d] += red[d + s];
    __syncthreads();
  }
  if (d == 0) enorm[k] = red[0];
}

// ---------------- Kernel B: distances via WMMA, argmin, quantize -----------
__global__ __launch_bounds__(256) void vq_main(const float* __restrict__ lat,
                                               const float* __restrict__ emb,
                                               const unsigned short* __restrict__ ehi,
                                               const unsigned short* __restrict__ elo,
                                               const float* __restrict__ enorm,
                                               float* __restrict__ out,
                                               int* __restrict__ inds,
                                               float* __restrict__ wgsum) {
  extern __shared__ char smem[];
  unsigned short* xhi = (unsigned short*)(smem + XHI_OFF);  // [128][264] bf16
  unsigned short* xlo = (unsigned short*)(smem + XLO_OFF);
  float* xnorm = (float*)(smem + XN_OFF);                   // [128]
  unsigned long long* minkey = (unsigned long long*)(smem + MK_OFF); // [128]

  const int tid = threadIdx.x;
  const int wg  = blockIdx.x;          // 0..511
  const int b   = wg >> 5;             // 16 batches
  const int h0  = (wg & 31) << 1;      // two h rows per WG

  const int w = tid & 63;
  const int q = tid >> 6;              // 0..3

  const int wv   = tid >> 5;      // wave 0..7 -> M tile
  const int lane = tid & 31;
  const int l15  = lane & 15;
  const int hi16 = lane >> 4;
  const int m0   = wv * 16;

  if (tid < 128) minkey[tid] = ~0ull;

  // Async prefetch of one 64-code chunk (hi+lo) into LDS buffer `buf`.
  // Per wave: 8 rows x 2 (hi/lo) = 16 global_load_async_to_lds_b128
  // (each wave-instruction moves one 512 B codebook row: 32 lanes x 16 B).
  auto issue_chunk = [&](int cc, int buf) {
    const unsigned ebase = (unsigned)(EB_OFF + buf * EB_STRIDE);
    #pragma unroll
    for (int r = 0; r < 8; ++r) {
      const int row = wv * 8 + r;
      const unsigned goff = (unsigned)((cc * 64 + row) * 512 + lane * 16);
      const unsigned lhi  = ebase + (unsigned)(row * ROWB + lane * 16);
      asm volatile("global_load_async_to_lds_b128 %0, %1, %2"
                   :: "v"(lhi), "v"(goff), "s"(ehi) : "memory");
      const unsigned llo = lhi + (unsigned)EHL_REL;
      asm volatile("global_load_async_to_lds_b128 %0, %1, %2"
                   :: "v"(llo), "v"(goff), "s"(elo) : "memory");
    }
  };

  // Kick off chunk 0 before staging X so the DMA overlaps the X conversion.
  issue_chunk(0, 0);

  // ---- stage latents tile: X[128 pixels][256 d] as bf16 hi/lo ----
  #pragma unroll 4
  for (int it = 0; it < 128; ++it) {
    const int pr = it * 4 + q;         // (d, h_local) pair index 0..511
    const int hl = pr & 1;
    const int d  = pr >> 1;
    const float v = lat[(((b * DDIM + d) * 64) + (h0 + hl)) * 64 + w];
    const unsigned short hb = f2bf_bits(v);
    const unsigned short lb = f2bf_bits(v - bf_bits2f(hb));
    const int p = hl * 64 + w;
    xhi[p * DPAD + d] = hb;
    xlo[p * DPAD + d] = lb;
  }
  __syncthreads();

  // ---- deterministic per-pixel ||x||^2 (fixed order) ----
  if (tid < 128) {
    float s = 0.f;
    #pragma unroll 8
    for (int d = 0; d < DDIM; ++d) {
      const float x = bf_bits2f(xhi[tid * DPAD + d]) + bf_bits2f(xlo[tid * DPAD + d]);
      s = fmaf(x, x, s);
    }
    xnorm[tid] = s;
  }
  __syncthreads();

  // Preload this lane's 8 row norms into registers (rows fixed per lane).
  float xn[8];
  #pragma unroll
  for (int r = 0; r < 8; ++r) xn[r] = xnorm[m0 + r + 8 * hi16];

  // A operand base (per lane): row m0+l15; halves of the 32-d chunk per hi16
  const char* aHi = smem + XHI_OFF + (m0 + l15) * ROWB + hi16 * 16;
  const char* aLo = smem + XLO_OFF + (m0 + l15) * ROWB + hi16 * 16;

  // Register-resident running min / argmin (8 rows per lane).
  float vmin[8];
  unsigned vidx[8];
  #pragma unroll
  for (int r = 0; r < 8; ++r) { vmin[r] = __builtin_inff(); vidx[r] = 0u; }

  for (int c = 0; c < 16; ++c) {       // 16 chunks of 64 codes
    const int buf = c & 1;
    // All waves done reading buf^1's previous chunk before we DMA into it.
    __syncthreads();
    if (c + 1 < 16) {
      issue_chunk(c + 1, buf ^ 1);
      // async loads complete in order: <=16 outstanding == chunk c landed
      asm volatile("s_wait_asynccnt 0x10" ::: "memory");
    } else {
      asm volatile("s_wait_asynccnt 0x0" ::: "memory");
    }
    __syncthreads();                   // publish chunk c to all waves

    const char* eb = smem + EB_OFF + buf * EB_STRIDE;

    // 4 live accumulators: load A once per d-step, reuse across 4 k-tiles.
    v8f acc[4] = {{}, {}, {}, {}};
    #pragma unroll
    for (int dstep = 0; dstep < 8; ++dstep) {    // 8 x 32-d steps
      const int off = dstep * 64;                // d0*2 bytes
      const v16bf A_hi = ld_pair(aHi + off, aHi + off + 32);
      const v16bf A_lo = ld_pair(aLo + off, aLo + off + 32);
      #pragma unroll
      for (int kt = 0; kt < 4; ++kt) {           // 4 k-tiles of 16 codes
        const char* bHi = eb + (kt * 16 + l15) * ROWB + hi16 * 16;
        const v16bf B_hi = ld_pair(bHi + off, bHi + off + 32);
        const v16bf B_lo = ld_pair(bHi + EHL_REL + off, bHi + EHL_REL + off + 32);
        acc[kt] = __builtin_amdgcn_wmma_f32_16x16x32_bf16(false, A_hi, false, B_hi,
                                                          (short)0, acc[kt], false, false);
        acc[kt] = __builtin_amdgcn_wmma_f32_16x16x32_bf16(false, A_hi, false, B_lo,
                                                          (short)0, acc[kt], false, false);
        acc[kt] = __builtin_amdgcn_wmma_f32_16x16x32_bf16(false, A_lo, false, B_hi,
                                                          (short)0, acc[kt], false, false);
      }
    }

    // Score update in registers (k ascending => argmin-first tie-break).
    #pragma unroll
    for (int kt = 0; kt < 4; ++kt) {
      const int kglob = c * 64 + kt * 16 + l15;
      const float en = enorm[kglob];
      #pragma unroll
      for (int r = 0; r < 8; ++r) {
        const float dv = fmaf(-2.0f, acc[kt][r], en);  // ||e||^2 - 2 x.e
        const bool lt = dv < vmin[r];
        vmin[r] = lt ? dv : vmin[r];
        vidx[r] = lt ? (unsigned)kglob : vidx[r];
      }
    }
  }

  // Cross-lane argmin: one packed ds_min_u64 per (lane,row).
  #pragma unroll
  for (int r = 0; r < 8; ++r) {
    const int row = m0 + r + 8 * hi16;           // C layout: vgpr r, lane half
    const float dist = xn[r] + vmin[r];          // full ||x-e||^2 >= 0
    const unsigned long long key =
        ((unsigned long long)__float_as_uint(dist) << 32) |
        (unsigned long long)vidx[r];
    atomicMin(&minkey[row], key);                // order-independent
  }
  __syncthreads();

  // ---- indices + deterministic distance-sum reduction ----
  float* red = (float*)(smem + EB_OFF);          // reuse E stage as scratch
  if (tid < 128) {
    const unsigned long long key = minkey[tid];
    const int idx = (int)(unsigned int)(key & 0xFFFFFFFFull);
    const float dist = __uint_as_float((unsigned int)(key >> 32));
    const int hl = tid >> 6, wl = tid & 63;
    inds[(b * 64 + h0 + hl) * 64 + wl] = idx;
    red[tid] = dist;
  }
  __syncthreads();
  #pragma unroll
  for (int s = 64; s > 0; s >>= 1) {
    if (tid < s) red[tid] += red[tid + s];
    __syncthreads();
  }
  if (tid == 0) wgsum[wg] = red[0];

  // ---- quantized output (== straight-through forward value), NCHW ----
  #pragma unroll 4
  for (int it = 0; it < 128; ++it) {
    const int pr = it * 4 + q;
    const int hl = pr & 1;
    const int d  = pr >> 1;
    const int p  = hl * 64 + w;
    const int idx = (int)(unsigned int)(minkey[p] & 0xFFFFFFFFull);
    out[(((b * DDIM + d) * 64) + (h0 + hl)) * 64 + w] = emb[idx * DDIM + d];
  }
}

// ---------------- Kernel C: deterministic loss finalize --------------------
__global__ void vq_loss_k(const float* __restrict__ wgsum, float* __restrict__ loss) {
  if (threadIdx.x == 0 && blockIdx.x == 0) {
    float s = 0.f;
    for (int i = 0; i < NWG; ++i) s += wgsum[i];
    // vq_loss = (BETA + ALPHA) * mean((quant-lat)^2) = 1.25 * sum(dist)/ (N*D)
    *loss = 1.25f * (s / (float)OUT_ELEMS);
  }
}

extern "C" void kernel_launch(void* const* d_in, const int* in_sizes, int n_in,
                              void* d_out, int out_size, void* d_ws, size_t ws_size,
                              hipStream_t stream) {
  const float* lat = (const float*)d_in[0];   // [16,256,64,64] f32
  const float* emb = (const float*)d_in[1];   // [1024,256] f32

  float* out  = (float*)d_out;                // [16,256,64,64]
  float* loss = out + OUT_ELEMS;              // scalar
  int*   inds = (int*)(out + OUT_ELEMS + 1);  // [16,64,64] int32

  char* ws = (char*)d_ws;
  unsigned short* ehi = (unsigned short*)ws;                 // 512 KB bf16 hi
  unsigned short* elo = (unsigned short*)(ws + 524288);      // 512 KB bf16 lo
  float* enorm        = (float*)(ws + 1048576);              // 4 KB
  float* wgsum        = (float*)(ws + 1052672);              // 2 KB

  vq_prep<<<KCODES, 256, 0, stream>>>(emb, ehi, elo, enorm);
  vq_main<<<NWG, 256, SMEM_BYTES, stream>>>(lat, emb, ehi, elo, enorm,
                                            out, inds, wgsum);
  vq_loss_k<<<1, 32, 0, stream>>>(wgsum, loss);
}
